// GroupConv_25598005084878
// MI455X (gfx1250) — matrix operand order repaired and checked by reading, not compiled, not run on hardware
//
#include <hip/hip_runtime.h>

typedef __attribute__((ext_vector_type(16))) _Float16 v16h;
typedef __attribute__((ext_vector_type(8)))  float    v8f;
typedef __attribute__((ext_vector_type(4)))  float    v4f;

#define O_CH  32
#define ROUT_ 16
#define NCH   512      // O_CH * ROUT_
#define KW    5
#define KK    25
#define KPAD  32
#define HH    128
#define WW    128
#define NTILE 32       // NCH / 16

// -------------------------------------------------------------------------
// Kernel 1: build rotated filter bank, emit f16 taps directly in the WMMA
// B-fragment layout (32x16 K-major tiles): element (k, n) of tile nt lives at
//   lane = n + 16*(k/16), slot = k%16
// Bfrag is [NTILE][32 lanes][16 slots] _Float16  (32 KB total in d_ws).
// -------------------------------------------------------------------------
__global__ void build_filters(const float* __restrict__ weight,
                              _Float16* __restrict__ Bfrag) {
    int oc = blockIdx.x * blockDim.x + threadIdx.x;
    if (oc >= NCH) return;
    int o = oc / ROUT_;
    int r = oc % ROUT_;
    const float* wp = weight + o * KK;           // I = RIN = 1 -> plane directly
    float theta = (6.283185307179586f / (float)ROUT_) * (float)r;
    float cs = cosf(theta), sn = sinf(theta);

    float tap[KPAD];
#pragma unroll
    for (int k = 0; k < KPAD; ++k) tap[k] = 0.0f;

    auto g = [&](int yy, int xx) -> float {
        if (yy < 0 || yy >= KW || xx < 0 || xx >= KW) return 0.0f;
        return wp[yy * KW + xx];
    };

    for (int ky = 0; ky < KW; ++ky) {
        for (int kx = 0; kx < KW; ++kx) {
            float xo = ((float)kx + 0.5f) * (2.0f / KW) - 1.0f;
            float yo = ((float)ky + 0.5f) * (2.0f / KW) - 1.0f;
            float xi =  cs * xo + sn * yo;
            float yi = -sn * xo + cs * yo;
            float px = (xi + 1.0f) * (KW * 0.5f) - 0.5f;
            float py = (yi + 1.0f) * (KW * 0.5f) - 0.5f;
            float fx = floorf(px), fy = floorf(py);
            int   ix = (int)fx,    iy = (int)fy;
            float wx = px - fx,    wy = py - fy;
            tap[ky * KW + kx] =
                g(iy,     ix    ) * (1.0f - wy) * (1.0f - wx) +
                g(iy,     ix + 1) * (1.0f - wy) * wx +
                g(iy + 1, ix    ) * wy * (1.0f - wx) +
                g(iy + 1, ix + 1) * wy * wx;
        }
    }

    int nt = oc / 16, n = oc % 16;
#pragma unroll
    for (int k = 0; k < KPAD; ++k) {
        int lane = n + 16 * (k / 16);
        int slot = k % 16;
        Bfrag[(size_t)nt * 512 + lane * 16 + slot] = (_Float16)tap[k];
    }
}

// -------------------------------------------------------------------------
// Kernel 2: implicit-GEMM conv via v_wmma_f32_16x16x32_f16.
// Block = 128 threads (4 wave32); each block covers 64 consecutive x of one
// (batch, row); each wave owns a 16-pixel M-tile and all 512 channels.
// -------------------------------------------------------------------------
__global__ void __launch_bounds__(128)
conv_wmma(const float* __restrict__ x,
          const _Float16* __restrict__ Bfrag,
          const float* __restrict__ bias,
          float* __restrict__ out) {
    __shared__ float sX[5 * 68];                       // input slab + halo
    __shared__ __align__(32) _Float16 sB[NTILE * 512]; // 32 KB B bank

    const int tid = threadIdx.x;
    const int g  = blockIdx.x;         // 0..4095
    const int xhalf = g & 1;           // two 64-wide halves per row
    const int y = (g >> 1) & 127;
    const int b = g >> 8;              // 16 batches
    const int xbase = xhalf * 64;

    // stage input slab: rows y-2..y+2, cols xbase-2..xbase+65 (zero-padded)
    for (int i = tid; i < 5 * 68; i += 128) {
        int ry = i / 68, rx = i % 68;
        int gy = y - 2 + ry, gx = xbase - 2 + rx;
        float v = 0.0f;
        if (gy >= 0 && gy < HH && gx >= 0 && gx < WW)
            v = x[(size_t)b * (HH * WW) + gy * WW + gx];
        sX[i] = v;
    }
    // stage full B fragment bank (2048 x uint4 = 32 KB)
    {
        const uint4* gB = (const uint4*)Bfrag;
        uint4* sB4 = (uint4*)sB;
#pragma unroll
        for (int i = 0; i < 16; ++i)
            sB4[tid + i * 128] = gB[tid + i * 128];
    }
    __syncthreads();

    const int w    = tid >> 5;         // wave 0..3
    const int lane = tid & 31;
    const int m    = lane & 15;        // pixel within M-tile
    const int h    = lane >> 4;        // lane half
    const int x0   = xbase + 16 * w;
    const int lxb  = 16 * w + m;       // local col at kx=0 (halo offset folds out)

    // A fragment, 16-bit A layout for 16x32 (contiguous, matches B bank):
    //  half 0: slots 0..15 -> K 0..15
    //  half 1: slots 0..15 -> K 16..31
    v16h a;
#pragma unroll
    for (int e = 0; e < 16; ++e) {
        int k = 16 * h + e;
        int ky = k / KW, kx = k % KW;
        float v = (k < KK) ? sX[ky * 68 + lxb + kx] : 0.0f;
        a[e] = (_Float16)v;
    }

    // c[j] holds pixel x0 + 8*h + j of channel oc = nt*16 + (lane&15)
    const size_t outbase =
        (size_t)b * NCH * (HH * WW) + (size_t)y * WW + (size_t)(x0 + 8 * h);

    for (int nt = 0; nt < NTILE; ++nt) {
        v16h bf = *(const v16h*)(sB + nt * 512 + lane * 16);
        v8f c = {};
        c = __builtin_amdgcn_wmma_f32_16x16x32_f16(
                /*neg_a=*/false, a, /*neg_b=*/false, bf,
                /*c_mod=*/(short)0, c, /*reuse_a=*/false, /*reuse_b=*/false);

        int oc = nt * 16 + (lane & 15);
        float bv = bias[oc];
        float* op = out + outbase + (size_t)oc * (HH * WW);
        v4f lo = { c[0] + bv, c[1] + bv, c[2] + bv, c[3] + bv };
        v4f hi = { c[4] + bv, c[5] + bv, c[6] + bv, c[7] + bv };
        // 512 MB streaming output >> 192 MB L2: non-temporal stores
        __builtin_nontemporal_store(lo, (v4f*)op);
        __builtin_nontemporal_store(hi, (v4f*)(op + 4));
    }
}

extern "C" void kernel_launch(void* const* d_in, const int* in_sizes, int n_in,
                              void* d_out, int out_size, void* d_ws, size_t ws_size,
                              hipStream_t stream) {
    const float* x      = (const float*)d_in[0];
    const float* weight = (const float*)d_in[1];
    const float* bias   = (const float*)d_in[2];
    float* out          = (float*)d_out;
    _Float16* Bfrag     = (_Float16*)d_ws;   // 32 KB fragment bank

    build_filters<<<2, 256, 0, stream>>>(weight, Bfrag);
    conv_wmma<<<4096, 128, 0, stream>>>(x, Bfrag, bias, out);
}
